// CMAE_8856222564944
// MI455X (gfx1250) — matrix operand order repaired and checked by reading, not compiled, not run on hardware
//
#include <hip/hip_runtime.h>
#include <hip/hip_bf16.h>

// ---------------------------------------------------------------------------
// Problem dims
// ---------------------------------------------------------------------------
#define BATCH   1024
#define NITEMS  50000
#define NPAD    50016   // NITEMS padded to multiple of 32 for GEMM2 tiling
#define K1      50000
#define K1PAD   50016   // K of GEMM1 padded to multiple of 32
#define H1DIM   600
#define H1PAD   608     // 600 padded to multiple of 32 (N of GEMM1, K of GEMM2)
#define H2DIM   200

typedef __attribute__((ext_vector_type(16))) _Float16 v16h;
typedef __attribute__((ext_vector_type(8)))  _Float16 v8h;
typedef __attribute__((ext_vector_type(8)))  float    v8f;

// ---------------------------------------------------------------------------
// WMMA helpers (CDNA5 wave32, v_wmma_f32_16x16x32_f16)
//
// A-fragment (16x32 f16, MxK): lane (0..15) holds row M=lane, lane 16..31 row
// M=lane-16. Per ISA layout: half-wave k-base kb = (lane>>4)*8; VGPR0..3 hold
// K=kb..kb+7, VGPR4..7 hold K=kb+16..kb+23  -> two contiguous 16B loads.
//
// B-fragment (32x16 f16, KxN): lane holds column N=lane&15; same K layout as A:
// kb = (lane>>4)*8; K=kb..kb+7 and kb+16..kb+23 -> two 16B loads from a
// column-major (i.e. row-major B^T) buffer.
// ---------------------------------------------------------------------------
__device__ __forceinline__ v16h load_a_frag(const _Float16* __restrict__ p) {
  v8h lo = *(const v8h*)(p);        // K = kb .. kb+7
  v8h hi = *(const v8h*)(p + 16);   // K = kb+16 .. kb+23
  return __builtin_shufflevector(lo, hi, 0,1,2,3,4,5,6,7,8,9,10,11,12,13,14,15);
}
__device__ __forceinline__ v16h load_b_frag(const _Float16* __restrict__ p) {
  v8h lo = *(const v8h*)(p);        // K = kb .. kb+7
  v8h hi = *(const v8h*)(p + 16);   // K = kb+16 .. kb+23
  return __builtin_shufflevector(lo, hi, 0,1,2,3,4,5,6,7,8,9,10,11,12,13,14,15);
}
__device__ __forceinline__ v8f wmma16(v16h a, v16h b, v8f c) {
  return __builtin_amdgcn_wmma_f32_16x16x32_f16(false, a, false, b,
                                                (short)0, c, false, false);
}

// ---------------------------------------------------------------------------
// Kernel 1: per-row L2 norm of x + convert to f16 into padded [BATCH][K1PAD]
// x row stays in L2 between the reduce and convert phases (regular TH).
// ---------------------------------------------------------------------------
__global__ __launch_bounds__(256)
void normalize_convert_kernel(const float* __restrict__ x,
                              _Float16* __restrict__ xn) {
  const int b = blockIdx.x;
  const int tid = threadIdx.x;
  const float* row = x + (size_t)b * K1;
  float s = 0.f;
  for (int k = tid; k < K1; k += 256) { float v = row[k]; s += v * v; }
  __shared__ float red[256];
  __shared__ float s_inv;
  red[tid] = s; __syncthreads();
  for (int w = 128; w > 0; w >>= 1) {
    if (tid < w) red[tid] += red[tid + w];
    __syncthreads();
  }
  if (tid == 0) s_inv = 1.0f / fmaxf(sqrtf(red[0]), 1e-12f);
  __syncthreads();
  const float inv = s_inv;
  _Float16* orow = xn + (size_t)b * K1PAD;
  for (int k = tid; k < K1PAD; k += 256)
    orow[k] = (k < K1) ? (_Float16)(row[k] * inv) : (_Float16)0.f;
}

// ---------------------------------------------------------------------------
// Kernel 2: W1 [K1][600] f32  ->  W1t [608][K1PAD] f16 (transposed, 0-padded)
// LDS-tiled; W1 is read-once -> non-temporal load so it does not evict the
// f16 working set (xn + W1t must stay L2-resident for GEMM1).
// ---------------------------------------------------------------------------
__global__ __launch_bounds__(256)
void w1_transpose_kernel(const float* __restrict__ W1,
                         _Float16* __restrict__ W1t) {
  __shared__ _Float16 tile[32][33];
  const int k0 = blockIdx.x * 32;
  const int n0 = blockIdx.y * 32;
  const int tx = threadIdx.x, ty = threadIdx.y;   // (32, 8)
  for (int i = ty; i < 32; i += 8) {
    int k = k0 + i, n = n0 + tx;
    float v = (k < K1 && n < H1DIM)
                ? __builtin_nontemporal_load(&W1[(size_t)k * H1DIM + n]) : 0.f;
    tile[i][tx] = (_Float16)v;
  }
  __syncthreads();
  for (int i = ty; i < 32; i += 8) {
    int n = n0 + i, k = k0 + tx;
    W1t[(size_t)n * K1PAD + k] = tile[tx][i];     // grid covers padded range
  }
}

// ---------------------------------------------------------------------------
// Kernel 3: GEMM1  h1 = tanh(xn @ W1 + b1)   [1024 x 608], K = 50016
// One wave per 32x32 tile, 2x2 WMMA micro-tile. unroll 2 keeps in-flight
// fragments + 32 accumulator VGPRs low enough to avoid allocator copies.
// ---------------------------------------------------------------------------
__global__ __launch_bounds__(32)
void gemm1_kernel(const _Float16* __restrict__ A, const _Float16* __restrict__ Bt,
                  const float* __restrict__ bias, float* __restrict__ H) {
  const int lane = threadIdx.x & 31;
  const int m0 = blockIdx.x * 32;
  const int n0 = blockIdx.y * 32;
  const int r  = lane & 15;
  const int ka = (lane >> 4) * 8;    // A half-wave k-base
  const int kb = (lane >> 4) * 8;    // B half-wave k-base
  const _Float16* a0 = A  + (size_t)(m0 + r) * K1PAD + ka;
  const _Float16* a1 = a0 + (size_t)16 * K1PAD;
  const _Float16* b0 = Bt + (size_t)(n0 + r) * K1PAD + kb;
  const _Float16* b1 = b0 + (size_t)16 * K1PAD;
  v8f c00 = {}, c01 = {}, c10 = {}, c11 = {};
#pragma unroll 2
  for (int k = 0; k < K1PAD; k += 32) {
    v16h af0 = load_a_frag(a0 + k);
    v16h af1 = load_a_frag(a1 + k);
    v16h bf0 = load_b_frag(b0 + k);
    v16h bf1 = load_b_frag(b1 + k);
    c00 = wmma16(af0, bf0, c00);
    c01 = wmma16(af0, bf1, c01);
    c10 = wmma16(af1, bf0, c10);
    c11 = wmma16(af1, bf1, c11);
  }
  const int cm = (lane >> 4) * 8;
  const int cn = lane & 15;
#pragma unroll
  for (int rr = 0; rr < 8; ++rr) {
    int m  = m0 + cm + rr;
    int nA = n0 + cn, nB = nA + 16;
    float bA = (nA < H1DIM) ? bias[nA] : 0.f;
    float bB = (nB < H1DIM) ? bias[nB] : 0.f;
    H[(size_t)m        * H1PAD + nA] = tanhf(c00[rr] + bA);
    H[(size_t)m        * H1PAD + nB] = tanhf(c01[rr] + bB);
    H[(size_t)(m + 16) * H1PAD + nA] = tanhf(c10[rr] + bA);
    H[(size_t)(m + 16) * H1PAD + nB] = tanhf(c11[rr] + bB);
  }
}

// ---------------------------------------------------------------------------
// Kernel 4: h2 = tanh(h1 @ W2 + b2)   [1024 x 200], tiny -> plain fp32
// ---------------------------------------------------------------------------
__global__ __launch_bounds__(256)
void mlp1_kernel(const float* __restrict__ H1v, const float* __restrict__ W2,
                 const float* __restrict__ b2, float* __restrict__ H2) {
  const int b = blockIdx.x, j = threadIdx.x;
  if (j >= H2DIM) return;
  const float* hrow = H1v + (size_t)b * H1PAD;
  float acc = b2[j];
  for (int k = 0; k < H1DIM; ++k) acc += hrow[k] * W2[(size_t)k * H2DIM + j];
  H2[(size_t)b * H2DIM + j] = tanhf(acc);
}

// ---------------------------------------------------------------------------
// Kernel 5: h3 = tanh(h2 @ W3 + b3) -> f16 [1024][608] (0-padded) + ||h3||^2
// ---------------------------------------------------------------------------
__global__ __launch_bounds__(608)
void mlp2_kernel(const float* __restrict__ H2v, const float* __restrict__ W3,
                 const float* __restrict__ b3, _Float16* __restrict__ H3h,
                 float* __restrict__ hsq) {
  const int b = blockIdx.x, n = threadIdx.x;  // n < 608
  float v = 0.f;
  if (n < H1DIM) {
    const float* hrow = H2v + (size_t)b * H2DIM;
    float acc = b3[n];
    for (int k = 0; k < H2DIM; ++k) acc += hrow[k] * W3[(size_t)k * H1DIM + n];
    v = tanhf(acc);
  }
  H3h[(size_t)b * H1PAD + n] = (_Float16)v;
  __shared__ float red[608];
  red[n] = v * v;
  __syncthreads();
  if (n < 96) red[n] += red[n + 512];
  __syncthreads();
  for (int w = 256; w > 0; w >>= 1) {
    if (n < w) red[n] += red[n + w];
    __syncthreads();
  }
  if (n == 0) hsq[b] = red[0];
}

// ---------------------------------------------------------------------------
// Kernel 6: item_emb -> f16 [NPAD][608] (0-padded) + ||e||^2 per item
// item_emb fp32 is read-once -> non-temporal load; the f16 copy is what we
// want resident in L2 for GEMM2.
// ---------------------------------------------------------------------------
__global__ __launch_bounds__(608)
void item_convert_kernel(const float* __restrict__ E, _Float16* __restrict__ Eh,
                         float* __restrict__ esq) {
  const int n = blockIdx.x, k = threadIdx.x;  // k < 608
  float v = (n < NITEMS && k < H1DIM)
              ? __builtin_nontemporal_load(&E[(size_t)n * H1DIM + k]) : 0.f;
  Eh[(size_t)n * H1PAD + k] = (_Float16)v;
  __shared__ float red[608];
  red[k] = v * v;
  __syncthreads();
  if (k < 96) red[k] += red[k + 512];
  __syncthreads();
  for (int w = 256; w > 0; w >>= 1) {
    if (k < w) red[k] += red[k + w];
    __syncthreads();
  }
  if (k == 0) esq[n] = red[0];
}

// ---------------------------------------------------------------------------
// Kernel 7: GEMM2  dist = ||h||^2 - 2 h@E^T + ||e||^2   [1024 x 50000]
// Same WMMA micro-kernel, K = 608 fully unrolled. dist is write-once and
// never re-read -> non-temporal stores keep L2 for item_f16.
// ---------------------------------------------------------------------------
__global__ __launch_bounds__(32)
void gemm2_kernel(const _Float16* __restrict__ A, const _Float16* __restrict__ Bt,
                  const float* __restrict__ hsq, const float* __restrict__ esq,
                  float* __restrict__ out) {
  const int lane = threadIdx.x & 31;
  const int m0 = blockIdx.x * 32;
  const int n0 = blockIdx.y * 32;
  const int r  = lane & 15;
  const int ka = (lane >> 4) * 8;
  const int kb = (lane >> 4) * 8;
  const _Float16* a0 = A  + (size_t)(m0 + r) * H1PAD + ka;
  const _Float16* a1 = a0 + (size_t)16 * H1PAD;
  const _Float16* b0 = Bt + (size_t)(n0 + r) * H1PAD + kb;
  const _Float16* b1 = b0 + (size_t)16 * H1PAD;
  v8f c00 = {}, c01 = {}, c10 = {}, c11 = {};
#pragma unroll
  for (int k = 0; k < H1PAD; k += 32) {
    v16h af0 = load_a_frag(a0 + k);
    v16h af1 = load_a_frag(a1 + k);
    v16h bf0 = load_b_frag(b0 + k);
    v16h bf1 = load_b_frag(b1 + k);
    c00 = wmma16(af0, bf0, c00);
    c01 = wmma16(af0, bf1, c01);
    c10 = wmma16(af1, bf0, c10);
    c11 = wmma16(af1, bf1, c11);
  }
  const int cm = (lane >> 4) * 8;
  const int cn = lane & 15;
#pragma unroll
  for (int rr = 0; rr < 8; ++rr) {
    int m  = m0 + cm + rr;
    int nA = n0 + cn, nB = nA + 16;
    float hA = hsq[m], hB = hsq[m + 16];
    if (nA < NITEMS) {
      float eA = esq[nA];
      __builtin_nontemporal_store(hA - 2.f * c00[rr] + eA,
                                  &out[(size_t)m        * NITEMS + nA]);
      __builtin_nontemporal_store(hB - 2.f * c10[rr] + eA,
                                  &out[(size_t)(m + 16) * NITEMS + nA]);
    }
    if (nB < NITEMS) {
      float eB = esq[nB];
      __builtin_nontemporal_store(hA - 2.f * c01[rr] + eB,
                                  &out[(size_t)m        * NITEMS + nB]);
      __builtin_nontemporal_store(hB - 2.f * c11[rr] + eB,
                                  &out[(size_t)(m + 16) * NITEMS + nB]);
    }
  }
}

// ---------------------------------------------------------------------------
// Launch: inputs = x, W1, b1, W2, b2, W3, b3, item_emb (all fp32)
// Workspace layout (peak ~168 MB; item_f16 aliases xn, which is dead
// after GEMM1 — stream order guarantees safety):
// ---------------------------------------------------------------------------
extern "C" void kernel_launch(void* const* d_in, const int* in_sizes, int n_in,
                              void* d_out, int out_size, void* d_ws, size_t ws_size,
                              hipStream_t stream) {
  (void)in_sizes; (void)n_in; (void)out_size; (void)ws_size;
  const float* x  = (const float*)d_in[0];
  const float* W1 = (const float*)d_in[1];
  const float* b1 = (const float*)d_in[2];
  const float* W2 = (const float*)d_in[3];
  const float* b2 = (const float*)d_in[4];
  const float* W3 = (const float*)d_in[5];
  const float* b3 = (const float*)d_in[6];
  const float* E  = (const float*)d_in[7];
  float* out = (float*)d_out;

  char* ws = (char*)d_ws;
  _Float16* xn    = (_Float16*)ws;                 // [BATCH][K1PAD]  f16 (102.4 MB)
  _Float16* itemh = (_Float16*)ws;                 // [NPAD][H1PAD]   f16 (60.8 MB, aliases xn)
  size_t off = (size_t)BATCH * K1PAD * sizeof(_Float16);
  _Float16* W1t = (_Float16*)(ws + off); off += (size_t)H1PAD * K1PAD * sizeof(_Float16);
  float*    H1v = (float*)(ws + off);    off += (size_t)BATCH * H1PAD * sizeof(float);
  float*    H2v = (float*)(ws + off);    off += (size_t)BATCH * H2DIM * sizeof(float);
  _Float16* H3h = (_Float16*)(ws + off); off += (size_t)BATCH * H1PAD * sizeof(_Float16);
  float*    hsq = (float*)(ws + off);    off += 4096;
  float*    esq = (float*)(ws + off);

  normalize_convert_kernel<<<BATCH, 256, 0, stream>>>(x, xn);
  w1_transpose_kernel<<<dim3(K1PAD / 32, H1PAD / 32), dim3(32, 8), 0, stream>>>(W1, W1t);
  gemm1_kernel<<<dim3(BATCH / 32, H1PAD / 32), 32, 0, stream>>>(xn, W1t, b1, H1v);
  mlp1_kernel<<<BATCH, 256, 0, stream>>>(H1v, W2, b2, H2v);
  mlp2_kernel<<<BATCH, 608, 0, stream>>>(H2v, W3, b3, H3h, hsq);
  item_convert_kernel<<<NPAD, 608, 0, stream>>>(E, itemh, esq);   // after gemm1 (alias)
  gemm2_kernel<<<dim3(BATCH / 32, NPAD / 32), 32, 0, stream>>>(H3h, itemh, hsq, esq, out);
}
